// RetrievalGuidedCompletion_82248623718829
// MI455X (gfx1250) — matrix operand order repaired and checked by reading, not compiled, not run on hardware
//
#include <hip/hip_runtime.h>
#include <hip/hip_bf16.h>

// Shapes from the reference
#define BDIM 64
#define SDIM 32
#define DDIM 512
#define NMEM 4096
#define KTOP 4
#define EPSV 1e-8f
#define XS   520   // padded LDS row stride in bf16 elems: 1040B rows walk the 64 banks

#define AS1 __attribute__((address_space(1)))
#define AS3 __attribute__((address_space(3)))

#if defined(__gfx1250__) && __has_builtin(__builtin_amdgcn_global_load_async_to_lds_b128) && \
    __has_builtin(__builtin_amdgcn_s_wait_asynccnt)
#define USE_ASYNC_LDS 1
#else
#define USE_ASYNC_LDS 0
#endif

typedef __attribute__((ext_vector_type(16))) __bf16 bf16x16;
typedef __attribute__((ext_vector_type(8)))  float  f32x8;
typedef __attribute__((ext_vector_type(4)))  int    v4i;

__device__ __forceinline__ unsigned short f2bf(float f) {
  unsigned u = __float_as_uint(f);
  u += 0x7fffu + ((u >> 16) & 1u);           // round-to-nearest-even
  return (unsigned short)(u >> 16);
}
__device__ __forceinline__ unsigned pack2bf(float a, float b) {
  unsigned ua = __float_as_uint(a); ua += 0x7fffu + ((ua >> 16) & 1u);
  unsigned ub = __float_as_uint(b); ub += 0x7fffu + ((ub >> 16) & 1u);
  return (ua >> 16) | (ub & 0xffff0000u);
}

// ---- K0: f32 -> bf16 weight conversion (two arrays at once) ------------------
__global__ void cvt_bf16_kernel(const float* __restrict__ a, const float* __restrict__ b,
                                unsigned* __restrict__ oa, unsigned* __restrict__ ob,
                                int npairs) {
  int i = blockIdx.x * blockDim.x + threadIdx.x;
  if (i < npairs) {
    oa[i] = pack2bf(a[2*i], a[2*i+1]);
    ob[i] = pack2bf(b[2*i], b[2*i+1]);
  }
}

// ---- K1: row means over S (HBM-bound scan of 512MB) --------------------------
// rows: [0,64) text->q0  [64,128) image->q1  [128,4224) mem_image->m0  [4224,8320) mem_text->m1
__global__ void means_kernel(const float* __restrict__ text, const float* __restrict__ image,
                             const float* __restrict__ mem_image, const float* __restrict__ mem_text,
                             float* __restrict__ q0, float* __restrict__ q1,
                             float* __restrict__ m0, float* __restrict__ m1) {
  int row = blockIdx.x;
  const float* src; float* dst;
  if (row < 64)            { src = text      + (size_t)row        * SDIM*DDIM; dst = q0 + (size_t)row        * DDIM; }
  else if (row < 128)      { src = image     + (size_t)(row-64)   * SDIM*DDIM; dst = q1 + (size_t)(row-64)   * DDIM; }
  else if (row < 128+NMEM) { src = mem_image + (size_t)(row-128)  * SDIM*DDIM; dst = m0 + (size_t)(row-128)  * DDIM; }
  else                     { src = mem_text  + (size_t)(row-4224) * SDIM*DDIM; dst = m1 + (size_t)(row-4224) * DDIM; }
  int c = threadIdx.x * 4;                       // 128 threads * float4 = 512 cols
  float4 acc = {0.f, 0.f, 0.f, 0.f};
  for (int s = 0; s < SDIM; ++s) {
    float4 v = *(const float4*)(src + (size_t)s * DDIM + c);
    acc.x += v.x; acc.y += v.y; acc.z += v.z; acc.w += v.w;
  }
  const float inv = 1.0f / 32.0f;
  float4 o = {acc.x*inv, acc.y*inv, acc.z*inv, acc.w*inv};
  *(float4*)(dst + c) = o;
}

// ---- K2: cosine similarity + top-4 (block per (dir,b); wave per memory row) --
__global__ void topk_kernel(const float* __restrict__ q0, const float* __restrict__ q1,
                            const float* __restrict__ m0, const float* __restrict__ m1,
                            int* __restrict__ idxOut) {
  __shared__ float qs[DDIM];
  __shared__ float red[256];
  __shared__ float tv[8][4];
  __shared__ int   ti[8][4];
  int dir = blockIdx.x >> 6, b = blockIdx.x & 63;
  const float* q  = (dir == 0 ? q0 : q1) + (size_t)b * DDIM;
  const float* mm = (dir == 0 ? m0 : m1);
  int tid = threadIdx.x, lane = tid & 31, wave = tid >> 5;

  for (int i = tid; i < DDIM; i += 256) qs[i] = q[i];
  __syncthreads();
  float qss = 0.f;
  for (int i = tid; i < DDIM; i += 256) { float v = qs[i]; qss += v * v; }
  red[tid] = qss; __syncthreads();
  for (int s = 128; s > 0; s >>= 1) { if (tid < s) red[tid] += red[tid + s]; __syncthreads(); }
  float qden = fmaxf(sqrtf(red[0]), EPSV);
  __syncthreads();

  float bv[4] = {-1e30f, -1e30f, -1e30f, -1e30f};
  int   bi[4] = {-1, -1, -1, -1};
  for (int n = wave; n < NMEM; n += 8) {            // increasing n => stable ties
    const float* r = mm + (size_t)n * DDIM;
    float dot = 0.f, ss = 0.f, sm = 0.f;
    for (int d = lane * 4; d < DDIM; d += 128) {
      float4 v  = *(const float4*)(r + d);
      float4 qv = *(const float4*)(qs + d);
      dot += v.x*qv.x + v.y*qv.y + v.z*qv.z + v.w*qv.w;
      ss  += v.x*v.x + v.y*v.y + v.z*v.z + v.w*v.w;
      sm  += v.x + v.y + v.z + v.w;
    }
    for (int msk = 16; msk >= 1; msk >>= 1) {
      dot += __shfl_xor(dot, msk);
      ss  += __shfl_xor(ss,  msk);
      sm  += __shfl_xor(sm,  msk);
    }
    float mden = fmaxf(sqrtf(ss), EPSV);
    float sim = (sm != 0.0f) ? (dot / (qden * mden)) : 0.0f;   // sim * valid
    if (sim > bv[3]) {                                         // strict > keeps earlier idx on ties
      int p = 3;
      while (p > 0 && sim > bv[p-1]) { bv[p] = bv[p-1]; bi[p] = bi[p-1]; --p; }
      bv[p] = sim; bi[p] = n;
    }
  }
  if (lane == 0) for (int j = 0; j < 4; ++j) { tv[wave][j] = bv[j]; ti[wave][j] = bi[j]; }
  __syncthreads();
  if (tid == 0) {
    for (int sel = 0; sel < 4; ++sel) {
      float best = -1e30f; int besti = 0x7fffffff, bw = 0, bj = 0;
      for (int w = 0; w < 8; ++w)
        for (int j = 0; j < 4; ++j) {
          float v = tv[w][j]; int ix = ti[w][j];
          if (ix >= 0 && (v > best || (v == best && ix < besti))) { best = v; besti = ix; bw = w; bj = j; }
        }
      idxOut[(dir * 64 + b) * 4 + sel] = besti;
      ti[bw][bj] = -1;
    }
  }
}

// ---- K3: routing scores. mean(linear(x)) == linear(mean(x)), so only row
// projections of qbar and gathered mbar rows are needed. Softmax over K=4. ----
__global__ void score_kernel(const float* __restrict__ q0, const float* __restrict__ q1,
                             const float* __restrict__ m0, const float* __restrict__ m1,
                             const int* __restrict__ idxArr,
                             const float* __restrict__ remW0, const float* __restrict__ remB0,
                             const float* __restrict__ retW0, const float* __restrict__ retB0,
                             const float* __restrict__ remW1, const float* __restrict__ remB1,
                             const float* __restrict__ retW1, const float* __restrict__ retB1,
                             float* __restrict__ scoreOut) {
  __shared__ float qs[DDIM];
  __shared__ float ms[4][DDIM];
  __shared__ float red[256];
  int dir = blockIdx.x >> 6, b = blockIdx.x & 63;
  const float* q    = (dir == 0 ? q0 : q1) + (size_t)b * DDIM;
  const float* mm   = (dir == 0 ? m0 : m1);
  const float* remW = dir == 0 ? remW0 : remW1;
  const float* remB = dir == 0 ? remB0 : remB1;
  const float* retW = dir == 0 ? retW0 : retW1;
  const float* retB = dir == 0 ? retB0 : retB1;
  int tid = threadIdx.x;

  for (int i = tid; i < DDIM; i += 256) qs[i] = q[i];
  for (int k = 0; k < 4; ++k) {
    int id = idxArr[(dir * 64 + b) * 4 + k];
    for (int i = tid; i < DDIM; i += 256) ms[k][i] = mm[(size_t)id * DDIM + i];
  }
  __syncthreads();

  float pd[4] = {0.f, 0.f, 0.f, 0.f};
  for (int e = tid * 2; e < tid * 2 + 2; ++e) {
    const float* wr = remW + (size_t)e * DDIM;
    const float* wt = retW + (size_t)e * DDIM;
    float rb = remB[e];
    float t0 = retB[e], t1 = retB[e], t2 = retB[e], t3 = retB[e];
    for (int d = 0; d < DDIM; ++d) {
      float a = wr[d], c = wt[d];
      rb += qs[d] * a;
      t0 += ms[0][d] * c; t1 += ms[1][d] * c; t2 += ms[2][d] * c; t3 += ms[3][d] * c;
    }
    pd[0] += rb * t0; pd[1] += rb * t1; pd[2] += rb * t2; pd[3] += rb * t3;
  }
  float dk[4];
  for (int k = 0; k < 4; ++k) {
    red[tid] = pd[k]; __syncthreads();
    for (int s = 128; s > 0; s >>= 1) { if (tid < s) red[tid] += red[tid + s]; __syncthreads(); }
    dk[k] = red[0]; __syncthreads();
  }
  if (tid == 0) {
    float mx = fmaxf(fmaxf(dk[0], dk[1]), fmaxf(dk[2], dk[3]));
    float e0 = expf(dk[0]-mx), e1 = expf(dk[1]-mx), e2 = expf(dk[2]-mx), e3 = expf(dk[3]-mx);
    float s = e0 + e1 + e2 + e3;
    float* so = scoreOut + (dir * 64 + b) * 4;
    so[0] = e0/s; so[1] = e1/s; so[2] = e2/s; so[3] = e3/s;
  }
}

// ---- K4: expert MLP with WMMA bf16. Block = (dir,b); 8 waves; wave owns
// mtile = wave&1, ntiles ngrp*8..+7. Experts accumulate into one persistent
// f32 accumulator by folding score into the ReLU output. X and H share one
// 33KB LDS buffer. X gather uses gfx1250 async global->LDS when available. ----
__global__ void __launch_bounds__(256)
expert_kernel(const float* __restrict__ memImg, const float* __restrict__ memTxt,
              const int* __restrict__ idxArr, const float* __restrict__ scoreArr,
              const unsigned short* __restrict__ w1bf, const unsigned short* __restrict__ w2bf,
              const float* __restrict__ eb1_0, const float* __restrict__ eb2_0,
              const float* __restrict__ eb1_1, const float* __restrict__ eb2_1,
              float* __restrict__ outBase) {
  __shared__ unsigned short Tl[SDIM][XS];   // X, then H (overlaid)
  __shared__ float scS[4];
#if USE_ASYNC_LDS
  __shared__ float Xstage[8 * DDIM];        // 16KB async staging (8 rows of f32)
#endif

  int dir = blockIdx.x >> 6, b = blockIdx.x & 63;
  const float* memSrc = dir == 0 ? memImg : memTxt;
  const float* eb1    = dir == 0 ? eb1_0  : eb1_1;
  const float* eb2    = dir == 0 ? eb2_0  : eb2_1;
  const unsigned short* W1base = w1bf + (size_t)dir * (4 * DDIM * DDIM);
  const unsigned short* W2base = w2bf + (size_t)dir * (4 * DDIM * DDIM);
  float* gen = outBase + (size_t)(2 + dir) * (BDIM * SDIM * DDIM) + (size_t)b * SDIM * DDIM;

  int tid = threadIdx.x, lane = tid & 31, wave = tid >> 5;
  int mtile = wave & 1, ngrp = wave >> 1;
  if (tid < 4) scS[tid] = scoreArr[(dir * 64 + b) * 4 + tid];

  const f32x8 zero8 = {0.f,0.f,0.f,0.f,0.f,0.f,0.f,0.f};
  f32x8 oacc[8];
#pragma unroll
  for (int j = 0; j < 8; ++j) oacc[j] = zero8;

  // A-matrix 16x32 bf16 layout: lanes0-15 m=lane hold K {0..7,16..23}; lanes16-31 {8..15,24..31}
  // B-matrix 32x16 bf16 layout: n = lane&15; lanes0-15 hold K {0..7,16..23}; lanes16-31 {8..15,24..31}
  const int mrow   = mtile * 16 + (lane & 15);
  const int aoff   = (lane < 16) ? 0 : 8;
  const int boff   = (lane < 16) ? 0 : 8;
  const int ncBase = ngrp * 128 + (lane & 15);
  const int rbase  = mtile * 16 + ((lane < 16) ? 0 : 8);  // C/D: vgpr r -> row rbase+r, col = n

  for (int k = 0; k < 4; ++k) {
    __syncthreads();                                     // prev GEMM2 done with Tl
    // gather X = memory[idx[b,k]] (32x512 f32) -> LDS bf16
    int id = idxArr[(dir * 64 + b) * 4 + k];
    const float4* src4 = (const float4*)(memSrc + (size_t)id * SDIM * DDIM);
#if USE_ASYNC_LDS
    for (int c = 0; c < 4; ++c) {                        // 8 rows (16KB) per chunk
      const float4* gsrc = src4 + c * (8 * DDIM / 4);
      for (int t = tid; t < 8 * DDIM / 4; t += 256) {
        __builtin_amdgcn_global_load_async_to_lds_b128(
            (AS1 v4i*)(gsrc + t), (AS3 v4i*)(Xstage + 4 * t), 0, 0);
      }
      __builtin_amdgcn_s_wait_asynccnt(0);               // my async stores to LDS done
      __syncthreads();                                   // everyone's done -> visible
      for (int t = tid; t < 8 * DDIM / 4; t += 256) {
        int s = t >> 7, c4 = t & 127;
        float4 v = *(const float4*)(Xstage + 4 * t);
        *(uint2*)&Tl[c * 8 + s][c4 * 4] = make_uint2(pack2bf(v.x, v.y), pack2bf(v.z, v.w));
      }
      __syncthreads();                                   // Xstage reusable
    }
#else
    for (int t = tid; t < SDIM * (DDIM / 4); t += 256) {
      int s = t >> 7, c4 = t & 127;
      float4 v = src4[t];
      *(uint2*)&Tl[s][c4 * 4] = make_uint2(pack2bf(v.x, v.y), pack2bf(v.z, v.w));
    }
    __syncthreads();                                     // X ready
#endif

    // GEMM1: H = X @ W1^T   (K = d, N = e; W1[e,d] rows are contiguous == B lanes)
    const unsigned short* W1 = W1base + (size_t)k * DDIM * DDIM;
    f32x8 hacc[8];
#pragma unroll
    for (int j = 0; j < 8; ++j) hacc[j] = zero8;
    for (int ks = 0; ks < 16; ++ks) {
      int d0 = ks * 32;
      union { bf16x16 v; uint4 q[2]; } A;
      A.q[0] = *(const uint4*)&Tl[mrow][d0 + aoff];
      A.q[1] = *(const uint4*)&Tl[mrow][d0 + aoff + 16];
#pragma unroll
      for (int j = 0; j < 8; ++j) {
        const unsigned short* wr = W1 + (size_t)(ncBase + j * 16) * DDIM + d0 + boff;
        __builtin_prefetch((const void*)(wr + 32), 0, 3);   // near-scope prefetch of next K-step
        union { bf16x16 v; uint4 q[2]; } Bf;
        Bf.q[0] = *(const uint4*)(wr);
        Bf.q[1] = *(const uint4*)(wr + 16);
        hacc[j] = __builtin_amdgcn_wmma_f32_16x16x32_bf16(
            false, A.v, false, Bf.v, (short)0, hacc[j], false, false);
      }
    }
    __syncthreads();                                     // all X reads done; Tl reusable
    // epilogue: +b1, ReLU, *score_k, store H (bf16) over X
    float sck = scS[k];
#pragma unroll
    for (int j = 0; j < 8; ++j) {
      int ncol = ncBase + j * 16;
      float b1 = eb1[k * DDIM + ncol];
#pragma unroll
      for (int r = 0; r < 8; ++r) {
        float v = hacc[j][r] + b1;
        v = v > 0.0f ? v * sck : 0.0f;
        Tl[rbase + r][ncol] = f2bf(v);
      }
    }
    __syncthreads();                                     // H ready

    // GEMM2: oacc += (score*H) @ W2^T   (K = e, N = f)
    const unsigned short* W2 = W2base + (size_t)k * DDIM * DDIM;
    for (int ks = 0; ks < 16; ++ks) {
      int e0 = ks * 32;
      union { bf16x16 v; uint4 q[2]; } A;
      A.q[0] = *(const uint4*)&Tl[mrow][e0 + aoff];
      A.q[1] = *(const uint4*)&Tl[mrow][e0 + aoff + 16];
#pragma unroll
      for (int j = 0; j < 8; ++j) {
        const unsigned short* wr = W2 + (size_t)(ncBase + j * 16) * DDIM + e0 + boff;
        __builtin_prefetch((const void*)(wr + 32), 0, 3);
        union { bf16x16 v; uint4 q[2]; } Bf;
        Bf.q[0] = *(const uint4*)(wr);
        Bf.q[1] = *(const uint4*)(wr + 16);
        oacc[j] = __builtin_amdgcn_wmma_f32_16x16x32_bf16(
            false, A.v, false, Bf.v, (short)0, oacc[j], false, false);
      }
    }
  }

  // final bias: sum_k score_k * eb2[k][f]; full-overwrite store of gen[b]
#pragma unroll
  for (int j = 0; j < 8; ++j) {
    int ncol = ncBase + j * 16;
    float b2 = scS[0] * eb2[0 * DDIM + ncol] + scS[1] * eb2[1 * DDIM + ncol]
             + scS[2] * eb2[2 * DDIM + ncol] + scS[3] * eb2[3 * DDIM + ncol];
#pragma unroll
    for (int r = 0; r < 8; ++r)
      gen[(size_t)(rbase + r) * DDIM + ncol] = oacc[j][r] + b2;
  }
}

// ---- K5: masking / completion. gen slots (2,3 of d_out) were written unmasked.
__global__ void finalize_kernel(const float* __restrict__ image, const float* __restrict__ text,
                                const int* __restrict__ m1, const int* __restrict__ m2,
                                float* __restrict__ out) {
  size_t i = (size_t)blockIdx.x * blockDim.x + threadIdx.x;
  const size_t M = (size_t)BDIM * SDIM * DDIM;
  if (i >= M) return;
  int b = (int)(i / (SDIM * DDIM));
  bool m1e = (m1[b] == 1), m2e = (m2[b] == 1);
  float gi = out[2 * M + i];
  float gt = out[3 * M + i];
  out[0 * M + i] = (!m1e && m2e) ? gi : image[i];
  out[1 * M + i] = (!m2e && m1e) ? gt : text[i];
  out[2 * M + i] = m2e ? gi : 0.0f;
  out[3 * M + i] = m1e ? gt : 0.0f;
}

extern "C" void kernel_launch(void* const* d_in, const int* in_sizes, int n_in,
                              void* d_out, int out_size, void* d_ws, size_t ws_size,
                              hipStream_t stream) {
  const float* image     = (const float*)d_in[0];
  const float* text      = (const float*)d_in[1];
  const int*   m1        = (const int*)d_in[2];
  const int*   m2        = (const int*)d_in[3];
  const float* mem_image = (const float*)d_in[4];
  const float* mem_text  = (const float*)d_in[5];
  const float* img_rem_w = (const float*)d_in[6];
  const float* img_rem_b = (const float*)d_in[7];
  const float* img_ret_w = (const float*)d_in[8];
  const float* img_ret_b = (const float*)d_in[9];
  const float* img_ew1   = (const float*)d_in[10];
  const float* img_eb1   = (const float*)d_in[11];
  const float* img_ew2   = (const float*)d_in[12];
  const float* img_eb2   = (const float*)d_in[13];
  const float* txt_rem_w = (const float*)d_in[14];
  const float* txt_rem_b = (const float*)d_in[15];
  const float* txt_ret_w = (const float*)d_in[16];
  const float* txt_ret_b = (const float*)d_in[17];
  const float* txt_ew1   = (const float*)d_in[18];
  const float* txt_eb1   = (const float*)d_in[19];
  const float* txt_ew2   = (const float*)d_in[20];
  const float* txt_eb2   = (const float*)d_in[21];

  char* ws = (char*)d_ws;
  size_t off = 0;
  auto alloc = [&](size_t bytes) -> void* {
    void* p = ws + off;
    off = (off + bytes + 255) & ~(size_t)255;
    return p;
  };
  float* mb0 = (float*)alloc((size_t)NMEM * DDIM * 4);   // mean(mem_image)
  float* mb1 = (float*)alloc((size_t)NMEM * DDIM * 4);   // mean(mem_text)
  float* q0  = (float*)alloc((size_t)BDIM * DDIM * 4);   // mean(text)  (query for dir0)
  float* q1  = (float*)alloc((size_t)BDIM * DDIM * 4);   // mean(image) (query for dir1)
  int*   idxArr   = (int*)  alloc(2 * BDIM * KTOP * 4);
  float* scoreArr = (float*)alloc(2 * BDIM * KTOP * 4);
  unsigned short* w1bf = (unsigned short*)alloc((size_t)2 * 4 * DDIM * DDIM * 2);
  unsigned short* w2bf = (unsigned short*)alloc((size_t)2 * 4 * DDIM * DDIM * 2);
  (void)ws_size; (void)in_sizes; (void)n_in; (void)out_size;

  const int npairs = 4 * DDIM * DDIM / 2;  // 524288 bf16 pairs per weight tensor
  cvt_bf16_kernel<<<(npairs + 255) / 256, 256, 0, stream>>>(
      img_ew1, txt_ew1, (unsigned*)w1bf, (unsigned*)(w1bf + (size_t)4 * DDIM * DDIM), npairs);
  cvt_bf16_kernel<<<(npairs + 255) / 256, 256, 0, stream>>>(
      img_ew2, txt_ew2, (unsigned*)w2bf, (unsigned*)(w2bf + (size_t)4 * DDIM * DDIM), npairs);

  means_kernel<<<128 + 2 * NMEM, 128, 0, stream>>>(text, image, mem_image, mem_text,
                                                   q0, q1, mb0, mb1);
  topk_kernel<<<2 * BDIM, 256, 0, stream>>>(q0, q1, mb0, mb1, idxArr);
  score_kernel<<<2 * BDIM, 256, 0, stream>>>(q0, q1, mb0, mb1, idxArr,
                                             img_rem_w, img_rem_b, img_ret_w, img_ret_b,
                                             txt_rem_w, txt_rem_b, txt_ret_w, txt_ret_b,
                                             scoreArr);
  expert_kernel<<<2 * BDIM, 256, 0, stream>>>(mem_image, mem_text, idxArr, scoreArr,
                                              w1bf, w2bf, img_eb1, img_eb2, txt_eb1, txt_eb2,
                                              (float*)d_out);
  finalize_kernel<<<(BDIM * SDIM * DDIM + 255) / 256, 256, 0, stream>>>(
      image, text, m1, m2, (float*)d_out);
}
